// CMoE_a_69355131896109
// MI455X (gfx1250) — compile-verified, matrix operand order checked by reading, not run on hardware
//
#include <hip/hip_runtime.h>
#include <hip/hip_bf16.h>
#include <math.h>
#include <stdint.h>

typedef __attribute__((ext_vector_type(16))) __bf16 v16bf;
typedef __attribute__((ext_vector_type(8)))  __bf16 v8bf;
typedef __attribute__((ext_vector_type(4)))  __bf16 v4bf;
typedef __attribute__((ext_vector_type(8)))  float  v8f;

#define TPB 256
#define BM  128
#define BN  128
#define KC  32
#define LDW 40                 // KC + 8 pad: row stride 80B = 20 banks -> conflict-free frags
#define LDS_BUF (BM * LDW)     // one 128x32(+pad) bf16 tile

// ---------------- init: zero output, invalidate slot->token map ----------------
__global__ void k_init(float* y, long long ny, int* tos, int ntos) {
  long long i = (long long)blockIdx.x * blockDim.x + threadIdx.x;
  long long stride = (long long)gridDim.x * blockDim.x;
  for (long long j = i; j < ny; j += stride) y[j] = 0.f;
  for (long long j = i; j < ntos; j += stride) tos[j] = -1;
}

// ---------------- routing: per-block expert histograms ----------------
__global__ void k_route_count(const int* __restrict__ tid, int* __restrict__ e_idx,
                              int* __restrict__ bc, int E) {
  __shared__ int hist[64];
  int t = threadIdx.x;
  if (t < E) hist[t] = 0;
  __syncthreads();
  int s = blockIdx.x * 128 + t;
  int e = (tid[s] % 5099) % E;
  e_idx[s] = e;
  atomicAdd(&hist[e], 1);
  __syncthreads();
  if (t < E) bc[blockIdx.x * E + t] = hist[t];
}

// ---------------- routing: exclusive scan over blocks, per expert ----------------
__global__ void k_route_scan(const int* __restrict__ bc, int* __restrict__ bb,
                             int NB, int E) {
  int e = threadIdx.x;
  int base = 0;
  for (int b = 0; b < NB; ++b) { bb[b * E + e] = base; base += bc[b * E + e]; }
}

// ---------------- routing: in-order slot assignment within each block ----------------
__global__ void k_route_pos(const int* __restrict__ e_idx, const int* __restrict__ bb,
                            int* __restrict__ pos, int E) {
  __shared__ int le[128];
  int t = threadIdx.x;
  int s0 = blockIdx.x * 128;
  for (int i = t; i < 128; i += blockDim.x) le[i] = e_idx[s0 + i];
  __syncthreads();
  int cnt = 0;
  int base = bb[blockIdx.x * E + t];   // thread t owns expert t -> deterministic order
  for (int i = 0; i < 128; ++i)
    if (le[i] == t) { pos[s0 + i] = base + cnt; ++cnt; }
}

// ---------------- dispatch: gather kept tokens to bf16 [E][cap][D] ----------------
__global__ void k_dispatch(const float* __restrict__ x, const int* __restrict__ e_idx,
                           const int* __restrict__ pos, __bf16* __restrict__ disp,
                           int* __restrict__ tos, int D, int cap) {
  int s = blockIdx.x;
  int p = pos[s];
  if (p >= cap) return;                 // capacity drop
  int e = e_idx[s];
  long long dst = ((long long)e * cap + p) * D;
  long long src = (long long)s * D;
  for (int i = threadIdx.x * 4; i < D; i += blockDim.x * 4) {
    float4 v = *(const float4*)(x + src + i);
    v4bf o; o[0] = (__bf16)v.x; o[1] = (__bf16)v.y; o[2] = (__bf16)v.z; o[3] = (__bf16)v.w;
    *(v4bf*)(disp + dst + i) = o;
  }
  if (threadIdx.x == 0) tos[e * cap + p] = s;
}

// ---------------- WMMA tile engine (double-buffered, async A path) ----------------
// CDNA5 16-bit A/B frag: lanes 0-15 hold K 0-7 & 16-23, lanes 16-31 hold K 8-15 & 24-31
__device__ __forceinline__ v16bf frag_ld(const __bf16* base, int row0, int lane) {
  int r  = row0 + (lane & 15);
  int kb = (lane >> 4) << 3;                 // 0 or 8
  const __bf16* p = base + r * LDW + kb;
  v8bf lo = *(const v8bf*)p;                 // K kb .. kb+7
  v8bf hi = *(const v8bf*)(p + 16);          // K kb+16 .. kb+23
  return __builtin_shufflevector(lo, hi, 0,1,2,3,4,5,6,7,8,9,10,11,12,13,14,15);
}

// A tile 128x32 bf16: 512 x 16B chunks, 2 async-to-LDS instructions per wave.
__device__ __forceinline__ void async_a_tile(const __bf16* __restrict__ A, long long lda,
                                             int k0, __bf16* Asb, int t) {
  #pragma unroll
  for (int i = 0; i < 2; ++i) {
    int idx = t + i * 256;
    int r = idx >> 2, seg = (idx & 3) * 8;
    unsigned long long ga = (unsigned long long)(uintptr_t)(A + (long long)r * lda + k0 + seg);
    unsigned lds = (unsigned)(uintptr_t)(Asb + r * LDW + seg);   // AS(3) offset in low 32b
    asm volatile("global_load_async_to_lds_b128 %0, %1, off"
                 :: "v"(lds), "v"(ga) : "memory");
  }
}

// B tile 128x32 f32: register prefetch (4 x float4 per thread)...
__device__ __forceinline__ void load_b_regs(float4* breg, const float* __restrict__ Bw,
                                            long long ldb, int k0, int t) {
  #pragma unroll
  for (int i = 0; i < 4; ++i) {
    int idx = t + i * 256;
    int r = idx >> 3, seg = (idx & 7) * 4;
    breg[i] = *(const float4*)(Bw + (long long)r * ldb + k0 + seg);
  }
}
// ...then fp32->bf16 convert + LDS store.
__device__ __forceinline__ void store_b_tile(const float4* breg, __bf16* Bsb, int t) {
  #pragma unroll
  for (int i = 0; i < 4; ++i) {
    int idx = t + i * 256;
    int r = idx >> 3, seg = (idx & 7) * 4;
    float4 w = breg[i];
    v4bf o; o[0] = (__bf16)w.x; o[1] = (__bf16)w.y; o[2] = (__bf16)w.z; o[3] = (__bf16)w.w;
    *(v4bf*)(Bsb + r * LDW + seg) = o;
  }
}

// C += A(128 x K, bf16, row-major lda) * B(128 x K, f32, row-major ldb)^T
__device__ __forceinline__ void mma_tiles(const __bf16* __restrict__ A, long long lda,
                                          const float* __restrict__ Bw, long long ldb,
                                          int K, v8f acc[2][4],
                                          __bf16* As2, __bf16* Bs2) {
  int t = threadIdx.x;
  int lane = t & 31, wid = t >> 5;
  int wm = (wid & 3) * 32;                   // 4 waves along M
  int wn = (wid >> 2) * 64;                  // 2 waves along N
  int nstep = K / KC;
  float4 breg[4];

  async_a_tile(A, lda, 0, As2, t);           // prologue -> buffer 0
  load_b_regs(breg, Bw, ldb, 0, t);

  for (int kt = 0; kt < nstep; ++kt) {
    int cur = kt & 1;
    __bf16* Asb = As2 + cur * LDS_BUF;
    __bf16* Bsb = Bs2 + cur * LDS_BUF;
    __syncthreads();                         // prior readers of this buffer pair are done
    store_b_tile(breg, Bsb, t);
    if (kt + 1 < nstep) {                    // prefetch next k-step into other buffer
      async_a_tile(A, lda, (kt + 1) * KC, As2 + (cur ^ 1) * LDS_BUF, t);
      load_b_regs(breg, Bw, ldb, (kt + 1) * KC, t);
      asm volatile("s_wait_asynccnt 0x2" ::: "memory");  // cur's 2 asyncs done (in-order)
    } else {
      asm volatile("s_wait_asynccnt 0x0" ::: "memory");
    }
    __syncthreads();                         // all waves' LDS writes visible
    v16bf a0 = frag_ld(Asb, wm,      lane);
    v16bf a1 = frag_ld(Asb, wm + 16, lane);
    #pragma unroll
    for (int j = 0; j < 4; ++j) {
      v16bf b = frag_ld(Bsb, wn + j * 16, lane);
      acc[0][j] = __builtin_amdgcn_wmma_f32_16x16x32_bf16(false, a0, false, b,
                                                          (short)0, acc[0][j], false, false);
      acc[1][j] = __builtin_amdgcn_wmma_f32_16x16x32_bf16(false, a1, false, b,
                                                          (short)0, acc[1][j], false, false);
    }
  }
  __syncthreads();
}

// ---------------- GEMM 1: H = relu(X * Wk^T)^2, bf16 out ----------------
__global__ void __launch_bounds__(TPB)
k_gemm_h(const __bf16* __restrict__ disp, const float* __restrict__ Wk,
         __bf16* __restrict__ H, int cap, int D, int F) {
  __shared__ __bf16 As[2 * LDS_BUF];
  __shared__ __bf16 Bs[2 * LDS_BUF];
  int e = blockIdx.z;
  int m0 = blockIdx.x * BM, n0 = blockIdx.y * BN;
  v8f acc[2][4];
  #pragma unroll
  for (int i = 0; i < 2; ++i)
    #pragma unroll
    for (int j = 0; j < 4; ++j)
      #pragma unroll
      for (int v = 0; v < 8; ++v) acc[i][j][v] = 0.f;

  mma_tiles(disp + ((long long)e * cap + m0) * D, D,
            Wk   + ((long long)e * F   + n0) * D, D, D, acc, As, Bs);

  int lane = threadIdx.x & 31, wid = threadIdx.x >> 5;
  int wm = (wid & 3) * 32, wn = (wid >> 2) * 64;
  int mh = (lane >> 4) << 3, nl = lane & 15;   // C/D frag: VGPR v -> M=v(+8), lane -> N
  #pragma unroll
  for (int i = 0; i < 2; ++i)
    #pragma unroll
    for (int j = 0; j < 4; ++j)
      #pragma unroll
      for (int v = 0; v < 8; ++v) {
        float rv = fmaxf(acc[i][j][v], 0.f);   // relu^2 via v_max + v_mul
        rv = rv * rv;
        int m = m0 + wm + i * 16 + v + mh;
        int n = n0 + wn + j * 16 + nl;
        H[((long long)e * cap + m) * F + n] = (__bf16)rv;
      }
}

// ---------------- GEMM 2: out = sigmoid(X*Wr^T) * (H*Wv^T), scatter to y ----------------
__global__ void __launch_bounds__(TPB)
k_gemm_out(const __bf16* __restrict__ disp, const __bf16* __restrict__ H,
           const float* __restrict__ Wr, const float* __restrict__ Wv,
           const int* __restrict__ tos, float* __restrict__ y,
           int cap, int D, int F) {
  __shared__ __bf16 As[2 * LDS_BUF];
  __shared__ __bf16 Bs[2 * LDS_BUF];
  int e = blockIdx.z;
  int m0 = blockIdx.x * BM, n0 = blockIdx.y * BN;
  v8f accR[2][4], accV[2][4];
  #pragma unroll
  for (int i = 0; i < 2; ++i)
    #pragma unroll
    for (int j = 0; j < 4; ++j)
      #pragma unroll
      for (int v = 0; v < 8; ++v) { accR[i][j][v] = 0.f; accV[i][j][v] = 0.f; }

  // R pre-activation: X(C x D) * Wr[e](D x D)^T, K = D
  mma_tiles(disp + ((long long)e * cap + m0) * D, D,
            Wr   + ((long long)e * D   + n0) * D, D, D, accR, As, Bs);
  // KV: H(C x F) * Wv[e](D x F)^T, K = F
  mma_tiles(H  + ((long long)e * cap + m0) * F, F,
            Wv + ((long long)e * D   + n0) * F, F, F, accV, As, Bs);

  int lane = threadIdx.x & 31, wid = threadIdx.x >> 5;
  int wm = (wid & 3) * 32, wn = (wid >> 2) * 64;
  int mh = (lane >> 4) << 3, nl = lane & 15;
  #pragma unroll
  for (int i = 0; i < 2; ++i)
    #pragma unroll
    for (int v = 0; v < 8; ++v) {
      int m = m0 + wm + i * 16 + v + mh;
      int tok = tos[e * cap + m];
      if (tok < 0) continue;                    // unclaimed slot / dropped -> y stays 0
      #pragma unroll
      for (int j = 0; j < 4; ++j) {
        float r = 1.f / (1.f + expf(-accR[i][j][v]));
        int n = n0 + wn + j * 16 + nl;
        y[(long long)tok * D + n] = r * accV[i][j][v];
      }
    }
}

extern "C" void kernel_launch(void* const* d_in, const int* in_sizes, int n_in,
                              void* d_out, int out_size, void* d_ws, size_t ws_size,
                              hipStream_t stream) {
  const float* x   = (const float*)d_in[0];
  const int*   tid = (const int*)  d_in[1];
  const float* Wk  = (const float*)d_in[2];
  const float* Wr  = (const float*)d_in[3];
  const float* Wv  = (const float*)d_in[4];
  float* y = (float*)d_out;

  long long S = in_sizes[1];                       // 32768
  long long D = in_sizes[0] / S;                   // 512
  long long E = (long long)in_sizes[3] / (D * D);  // 64
  long long F = (long long)in_sizes[2] / (E * D);  // 1792
  long long cap = (S + E - 1) / E; if (cap < 4) cap = 4;   // 512
  int NB = (int)(S / 128);                         // 256 routing blocks

  // carve workspace (~151 MB): routing arrays + bf16 dispatched + bf16 H
  char* p = (char*)d_ws;
  auto carve = [&](size_t bytes) { char* r = p; p += (bytes + 255) & ~(size_t)255; return r; };
  int*    e_idx = (int*)   carve((size_t)S * 4);
  int*    pos   = (int*)   carve((size_t)S * 4);
  int*    bc    = (int*)   carve((size_t)NB * E * 4);
  int*    bb    = (int*)   carve((size_t)NB * E * 4);
  int*    tos   = (int*)   carve((size_t)E * cap * 4);
  __bf16* disp  = (__bf16*)carve((size_t)E * cap * D * 2);
  __bf16* H     = (__bf16*)carve((size_t)E * cap * F * 2);

  k_init<<<1024, 256, 0, stream>>>(y, S * D, tos, (int)(E * cap));
  k_route_count<<<NB, 128, 0, stream>>>(tid, e_idx, bc, (int)E);
  k_route_scan<<<1, (int)E, 0, stream>>>(bc, bb, NB, (int)E);
  k_route_pos<<<NB, (int)E, 0, stream>>>(e_idx, bb, pos, (int)E);
  k_dispatch<<<(int)S, 128, 0, stream>>>(x, e_idx, pos, disp, tos, (int)D, (int)cap);

  dim3 gh((unsigned)(cap / BM), (unsigned)(F / BN), (unsigned)E);   // 4 x 14 x 64
  k_gemm_h<<<gh, TPB, 0, stream>>>(disp, Wk, H, (int)cap, (int)D, (int)F);

  dim3 go((unsigned)(cap / BM), (unsigned)(D / BN), (unsigned)E);   // 4 x 4 x 64
  k_gemm_out<<<go, TPB, 0, stream>>>(disp, H, Wr, Wv, tos, y, (int)cap, (int)D, (int)F);
}
